// InteractionAwareEquivariantBlock_420906795767
// MI455X (gfx1250) — matrix operand order repaired and checked by reading, not compiled, not run on hardware
//
#include <hip/hip_runtime.h>

typedef __attribute__((ext_vector_type(16))) _Float16 v16h;
typedef __attribute__((ext_vector_type(8)))  float    v8f;

static constexpr int ND = 64;   // NODE_DIM == HIDDEN_DIM

__device__ __forceinline__ float fast_sig(float x) {
  // 1/(1+e^-x) via v_exp_f32 + v_rcp_f32
  return __builtin_amdgcn_rcpf(1.0f + __builtin_amdgcn_exp2f(-1.4426950408889634f * x));
}
__device__ __forceinline__ float fast_silu(float x) { return x * fast_sig(x); }

__global__ __launch_bounds__(256)
void iaeb_kernel(const float* __restrict__ h_l, const float* __restrict__ x_l,
                 const float* __restrict__ h_r, const float* __restrict__ x_r,
                 const float* __restrict__ W1,  const float* __restrict__ b1,
                 const float* __restrict__ W2,  const float* __restrict__ b2,
                 const float* __restrict__ Wn1, const float* __restrict__ bn1,
                 const float* __restrict__ ln_g,const float* __restrict__ ln_b,
                 const float* __restrict__ Wn2, const float* __restrict__ bn2,
                 float* __restrict__ out, int N, int M)
{
  __shared__ float4 w1pack[64];        // {W1[0][k],W1[1][k],W1[2][k],W1[3][k]+b1[k]}
  __shared__ float  hrT[2][16][68];    // double-buffered h_r chunk, padded rows
  __shared__ float  wsc[8][64];        // per-wave scratch (h_agg, then silu(z))

  const int tid  = threadIdx.x;
  const int wv   = tid >> 5;           // wave in block (wave32)
  const int lane = tid & 31;
  const int half = lane >> 4;
  const int l16  = lane & 15;
  const int n    = blockIdx.x * 8 + wv;

  if (tid < 64)
    w1pack[tid] = make_float4(W1[tid], W1[64 + tid], W1[128 + tid],
                              W1[192 + tid] + b1[tid]);
  __syncthreads();

  // ---- W2 as WMMA B fragments: 4 column tiles x 2 K-halves.
  // B (32x16, f16): lanes 0-15 hold K=0..15 (elem e -> k=e), lanes 16-31 K=16..31.
  v16h bf[4][2];
#pragma unroll
  for (int t = 0; t < 4; ++t)
#pragma unroll
    for (int j = 0; j < 2; ++j)
#pragma unroll
      for (int e = 0; e < 16; ++e) {
        int k = 32 * j + 16 * half + e;
        bf[t][j][e] = (_Float16)W2[k * ND + (16 * t + l16)];
      }

  // ---- per-n invariants
  const float xl0 = x_l[n * 3 + 0], xl1 = x_l[n * 3 + 1], xl2 = x_l[n * 3 + 2];
  const float lp0 = h_l[n * ND + 61], lp1 = h_l[n * ND + 62], lp2 = h_l[n * ND + 63];
  float b2v[4];
#pragma unroll
  for (int t = 0; t < 4; ++t) b2v[t] = b2[16 * t + l16];

  float acc[4] = {0, 0, 0, 0}, wsum[4] = {0, 0, 0, 0};

  for (int m0 = 0, it = 0; m0 < M; m0 += 16, ++it) {
    const int buf = it & 1;
    {   // cooperative stage: 16x64 h_r tile, one b128 per thread
      int r = tid >> 4, c = (tid & 15) * 4;
      float4 v = *(const float4*)(h_r + (size_t)(m0 + r) * ND + c);
      *(float4*)(&hrT[buf][r][c]) = v;
    }
    __syncthreads();

    // edge features for row m = m0 + l16 (same for both lane halves, as A-layout needs)
    const int m = m0 + l16;
    float xr0 = x_r[m * 3 + 0], xr1 = x_r[m * 3 + 1], xr2 = x_r[m * 3 + 2];
    float rp0 = hrT[buf][l16][61], rp1 = hrT[buf][l16][62], rp2 = hrT[buf][l16][63];
    float dx = xl0 - xr0, dy = xl1 - xr1, dz = xl2 - xr2;
    float d2 = dx * dx + dy * dy + dz * dz;
    float dist = __builtin_amdgcn_exp2f(d2 * (-1.4426950408889634f / 5.0f));
    float hb = lp0 * rp1 + lp1 * rp0;
    float hy = lp2 * rp2;

    // hidden = silu(edge_raw @ W1 + b1) packed straight into A fragments
    // A (16x32, f16): elem e of lane -> k = 16*half + e
    v16h a0, a1;
#pragma unroll
    for (int e = 0; e < 16; ++e) {
      int k0 = 16 * half + e;
      float4 p = w1pack[k0];
      float4 q = w1pack[k0 + 32];
      float hA = dist * p.x + hb * p.y + hy * p.z + p.w;
      float hB = dist * q.x + hb * q.y + hy * q.z + q.w;
      a0[e] = (_Float16)fast_silu(hA);
      a1[e] = (_Float16)fast_silu(hB);
    }

    // e-tile = hidden @ W2 (+b2), sigmoid, weighted aggregation
#pragma unroll
    for (int t = 0; t < 4; ++t) {
      v8f c = {};
      c = __builtin_amdgcn_wmma_f32_16x16x32_f16(false, a0, false, bf[t][0],
                                                 (short)0, c, false, false);
      c = __builtin_amdgcn_wmma_f32_16x16x32_f16(false, a1, false, bf[t][1],
                                                 (short)0, c, false, false);
      const int col = 16 * t + l16;
#pragma unroll
      for (int d = 0; d < 8; ++d) {                 // D row = d + 8*half
        float w = fast_sig(c[d] + b2v[t]);
        float hr = hrT[buf][d + 8 * half][col];
        acc[t]  += w * hr;
        wsum[t] += w;
      }
    }
  }

  // ---- combine lane halves, finish aggregation
  float agg[4];
#pragma unroll
  for (int t = 0; t < 4; ++t) {
    float a = acc[t] + __shfl_xor(acc[t], 16, 32);
    float s = wsum[t] + __shfl_xor(wsum[t], 16, 32);
    agg[t] = a / (s + 1e-6f);
  }
  if (half == 0) {
#pragma unroll
    for (int t = 0; t < 4; ++t) wsc[wv][16 * t + l16] = agg[t];
  }
  __builtin_amdgcn_wave_barrier();

  // ---- node MLP 1: z = concat(h_l[n], h_agg) @ Wn1 + bn1   (lane owns o=2*lane, 2*lane+1)
  const float2* Wn1v = (const float2*)Wn1;
  float2 z = ((const float2*)bn1)[lane];
#pragma unroll 4
  for (int i = 0; i < 64; ++i) {
    float hl = h_l[n * ND + i];          // wave-uniform
    float ha = wsc[wv][i];
    float2 wa = Wn1v[i * 32 + lane];
    float2 wb = Wn1v[(64 + i) * 32 + lane];
    z.x += hl * wa.x + ha * wb.x;
    z.y += hl * wa.y + ha * wb.y;
  }

  // ---- LayerNorm over 64 channels (wave-wide butterfly reduction)
  float s = z.x + z.y, sq = z.x * z.x + z.y * z.y;
#pragma unroll
  for (int off = 1; off < 32; off <<= 1) {
    s  += __shfl_xor(s, off, 32);
    sq += __shfl_xor(sq, off, 32);
  }
  float mean = s * (1.0f / 64.0f);
  float var  = sq * (1.0f / 64.0f) - mean * mean;
  float rstd = __builtin_amdgcn_rsqf(var + 1e-5f);
  float2 g  = ((const float2*)ln_g)[lane];
  float2 be = ((const float2*)ln_b)[lane];
  float zn0 = (z.x - mean) * rstd * g.x + be.x;
  float zn1 = (z.y - mean) * rstd * g.y + be.y;
  wsc[wv][2 * lane]     = fast_silu(zn0);
  wsc[wv][2 * lane + 1] = fast_silu(zn1);
  __builtin_amdgcn_wave_barrier();

  // ---- node MLP 2 + residual
  const float2* Wn2v = (const float2*)Wn2;
  float2 o = ((const float2*)bn2)[lane];
#pragma unroll 4
  for (int i = 0; i < 64; ++i) {
    float sv = wsc[wv][i];
    float2 w2 = Wn2v[i * 32 + lane];
    o.x += sv * w2.x;
    o.y += sv * w2.y;
  }
  float2 hlv = ((const float2*)(h_l + (size_t)n * ND))[lane];
  float2 res; res.x = hlv.x + o.x; res.y = hlv.y + o.y;
  ((float2*)(out + (size_t)n * ND))[lane] = res;
}

extern "C" void kernel_launch(void* const* d_in, const int* in_sizes, int n_in,
                              void* d_out, int out_size, void* d_ws, size_t ws_size,
                              hipStream_t stream) {
  const float* h_l  = (const float*)d_in[0];
  const float* x_l  = (const float*)d_in[1];
  const float* h_r  = (const float*)d_in[2];
  const float* x_r  = (const float*)d_in[3];
  const float* W1   = (const float*)d_in[4];
  const float* b1   = (const float*)d_in[5];
  const float* W2   = (const float*)d_in[6];
  const float* b2   = (const float*)d_in[7];
  const float* Wn1  = (const float*)d_in[8];
  const float* bn1  = (const float*)d_in[9];
  const float* ln_g = (const float*)d_in[10];
  const float* ln_b = (const float*)d_in[11];
  const float* Wn2  = (const float*)d_in[12];
  const float* bn2  = (const float*)d_in[13];
  float* out = (float*)d_out;

  const int N = in_sizes[0] / ND;   // 2048
  const int M = in_sizes[2] / ND;   // 2048

  dim3 grid(N / 8), block(256);     // 8 wave32 per block, one wave per left node
  iaeb_kernel<<<grid, block, 0, stream>>>(h_l, x_l, h_r, x_r, W1, b1, W2, b2,
                                          Wn1, bn1, ln_g, ln_b, Wn2, bn2,
                                          out, N, M);
}
